// WaveletBlock_68616397521248
// MI455X (gfx1250) — compile-verified, matrix operand order checked by reading, not run on hardware
//
#include <hip/hip_runtime.h>
#include <hip/hip_bf16.h>

typedef __attribute__((ext_vector_type(16))) __bf16 v16bf;
typedef __attribute__((ext_vector_type(8)))  __bf16 v8bf;
typedef __attribute__((ext_vector_type(4)))  __bf16 v4bf;
typedef __attribute__((ext_vector_type(8)))  float  v8f;
typedef __attribute__((ext_vector_type(4)))  float  v4f;
typedef __attribute__((ext_vector_type(4)))  unsigned int u32x4;
typedef __attribute__((ext_vector_type(8)))  int    i32x8;
typedef __attribute__((ext_vector_type(4)))  int    i32x4;

// Problem constants
#define C64   64          // channels
#define NIMG  16          // B*L
#define LDK   264         // padded bf16 K-stride (264*2B = 528B -> 4-bank skew/row)
#define LDY   264         // padded f32 stride for y buffer

// LDS carve-up (bytes):
//   [0,       135168) : Wlds 256 x 264 bf16   (W1 then W2, DMA'd by TDM w/ padding)
//   [135168,  202752) : Bmat 128 x 264 bf16   (x_dwt tile, B-layout [col][K])
//   [202752,  270336) : Feat 128 x 264 bf16   (silu output, B-layout [col][K])
//   y fp32 128 x 264 = 135168 B reuses [135168, 270336)
#define LDS_BYTES 270336

// ---------------------------------------------------------------------------
// Pre-kernel: convert both 256x256 fp32 weight matrices to bf16 in d_ws.
// Layout: ws[0..65535] = W1 (row-major, stride 256), ws[65536..131071] = W2.
// ---------------------------------------------------------------------------
__global__ void convert_weights(const float* __restrict__ w1,
                                const float* __restrict__ w2,
                                __bf16* __restrict__ ws)
{
    int i = blockIdx.x * 256 + threadIdx.x;          // 0..32767 float4 tasks
    const float4* src = (i < 16384) ? (const float4*)w1 : (const float4*)w2;
    float4 f = src[i & 16383];
    v4bf v; v[0]=(__bf16)f.x; v[1]=(__bf16)f.y; v[2]=(__bf16)f.z; v[3]=(__bf16)f.w;
    *(v4bf*)&ws[(size_t)i * 4] = v;
}

// ---------------------------------------------------------------------------
// TDM: DMA one 256x256 bf16 weight matrix (row stride 256) from global into
// LDS at byte offset 0, inserting 16B of padding after every 512B row so the
// LDS image has the 264-element padded stride the WMMA fragment loads expect.
// ---------------------------------------------------------------------------
__device__ __forceinline__ void tdm_load_weight(const __bf16* gsrc)
{
    unsigned long long ga = (unsigned long long)(const void*)gsrc;
    u32x4 g0;
    g0[0] = 1u;                                   // count=1 (valid user D#)
    g0[1] = 0u;                                   // lds_addr = 0 (Wlds region)
    g0[2] = (unsigned)(ga & 0xFFFFFFFFu);         // global_addr[31:0]
    g0[3] = (unsigned)((ga >> 32) & 0x01FFFFFFu)  // global_addr[56:32]
          | (2u << 30);                           // type = 2 ("image")
    i32x8 g1;
    g1[0] = (1 << 16)     // data_size = 2 bytes
          | (1 << 20)     // pad_enable
          | (6 << 22)     // pad_interval: 128 dwords (one 512B row)
          | (3 << 25);    // pad_amount: 4 dwords (16B)
    g1[1] = (256 << 16);  // tensor_dim0[15:0] in bits 63:48
    g1[2] = (256 << 16);  // bits95:64 : tensor_dim0 hi=0 | tensor_dim1 lo=256
    g1[3] = (256 << 16);  // bits127:96: tensor_dim1 hi=0 | tile_dim0=256
    g1[4] = 256;          // bits159:128: tile_dim1=256 | tile_dim2=0
    g1[5] = 256;          // bits191:160: tensor_dim0_stride = 256 elements
    g1[6] = 0;            // stride0 hi | tensor_dim1_stride lo (unused, 2D)
    g1[7] = 0;
    i32x4 z4 = {0, 0, 0, 0};
#if defined(__clang_major__) && (__clang_major__ >= 23)
    i32x8 z8 = {0, 0, 0, 0, 0, 0, 0, 0};
    __builtin_amdgcn_tensor_load_to_lds(g0, g1, z4, z4, z8, 0);
#else
    __builtin_amdgcn_tensor_load_to_lds(g0, g1, z4, z4, 0);
#endif
}

__global__ __launch_bounds__(256, 1)
void wavelet_fused(const float* __restrict__ x,
                   const __bf16* __restrict__ wbf,   // d_ws: W1 then W2 (bf16)
                   const float* __restrict__ b1,
                   const float* __restrict__ b2,
                   float* __restrict__ out)
{
    extern __shared__ char lds[];
    __bf16* Wlds = (__bf16*)lds;                 // 256*264
    __bf16* Bmat = (__bf16*)(lds + 135168);      // 128*264
    __bf16* Feat = (__bf16*)(lds + 202752);      // 128*264
    float*  Ybuf = (float*) (lds + 135168);      // 128*264 f32 (reuse)

    const int tid  = threadIdx.x;
    const int lane = tid & 31;
    const int wv   = tid >> 5;          // wave 0..7
    const int half = lane >> 4;         // lane group
    const int r    = lane & 15;

    const int n    = blockIdx.x >> 5;   // image 0..15
    const int tile = blockIdx.x & 31;   // 128-col tile within 64x64 plane

    // ---- Kick off the W1 DMA (wave 0 only), overlapped with the DWT ------
    if (wv == 0) {
        tdm_load_weight(wbf);           // W1 -> Wlds (padded rows)
    }

    // ---------------- P0: DWT of x tile -> Bmat ([col][K] bf16) ------------
    {
        const float* xb = x + (size_t)n * (C64 * 128 * 128);
        #pragma unroll 2
        for (int it = 0; it < 32; ++it) {
            int i  = it * 256 + tid;             // 8192 = 64c * 128cols
            int c  = i >> 7;
            int t  = i & 127;
            int p  = tile * 128 + t;
            int hh = p >> 6, ww = p & 63;
            const float* px = xb + ((size_t)c * 128 + 2 * hh) * 128 + 2 * ww;
            float2 f0 = *(const float2*)px;          // row 2hh:   x1 x3
            float2 f1 = *(const float2*)(px + 128);  // row 2hh+1: x2 x4
            float x1 = 0.5f * f0.x, x3 = 0.5f * f0.y;
            float x2 = 0.5f * f1.x, x4 = 0.5f * f1.y;
            __bf16* bp = &Bmat[t * LDK + c];
            bp[0]   = (__bf16)( x1 + x2 + x3 + x4);
            bp[64]  = (__bf16)(-x1 - x2 + x3 + x4);
            bp[128] = (__bf16)(-x1 + x2 - x3 + x4);
            bp[192] = (__bf16)( x1 - x2 - x3 + x4);
        }
    }
    if (wv == 0) {
        __builtin_amdgcn_s_wait_tensorcnt(0);    // W1 resident in LDS
    }
    __syncthreads();

    const int col0 = wv * 16;                    // this wave's 16 columns

    // ---------------- P1: feat = silu(W1 @ x_dwt + b1) --------------------
    {
        v16bf bfrag[8];
        const __bf16* bb = &Bmat[(col0 + r) * LDK + 8 * half];
        #pragma unroll
        for (int k = 0; k < 8; ++k) {
            v8bf lo = *(const v8bf*)(bb + k * 32);
            v8bf hi = *(const v8bf*)(bb + k * 32 + 16);
            v16bf f;
            #pragma unroll
            for (int j = 0; j < 8; ++j) { f[j] = lo[j]; f[8 + j] = hi[j]; }
            bfrag[k] = f;
        }
        for (int m = 0; m < 16; ++m) {
            v8f acc;
            float4 bl = *(const float4*)&b1[m * 16 + 8 * half];
            float4 bh = *(const float4*)&b1[m * 16 + 8 * half + 4];
            acc[0]=bl.x; acc[1]=bl.y; acc[2]=bl.z; acc[3]=bl.w;
            acc[4]=bh.x; acc[5]=bh.y; acc[6]=bh.z; acc[7]=bh.w;
            const __bf16* aa = &Wlds[(m * 16 + r) * LDK + 8 * half];
            #pragma unroll
            for (int k = 0; k < 8; ++k) {
                v8bf lo = *(const v8bf*)(aa + k * 32);
                v8bf hi = *(const v8bf*)(aa + k * 32 + 16);
                v16bf af;
                #pragma unroll
                for (int j = 0; j < 8; ++j) { af[j] = lo[j]; af[8 + j] = hi[j]; }
                acc = __builtin_amdgcn_wmma_f32_16x16x32_bf16(
                          false, af, false, bfrag[k], (short)0, acc, false, false);
            }
            v8bf fb;
            #pragma unroll
            for (int j = 0; j < 8; ++j) {
                float v = acc[j];
                // silu via fast v_rcp_f32 (bf16 output: full div is wasted work)
                float s = v * __builtin_amdgcn_rcpf(1.0f + __expf(-v));
                fb[j] = (__bf16)s;
            }
            // store into next GEMM's B layout: Feat[col][k = m*16 + 8*half + j]
            *(v8bf*)&Feat[(col0 + r) * LDK + m * 16 + 8 * half] = fb;
        }
    }
    __syncthreads();                             // all waves done reading W1

    // ---- DMA W2 over Wlds (wave 0 only) -----------------------------------
    if (wv == 0) {
        tdm_load_weight(wbf + 65536);            // W2 -> Wlds
        __builtin_amdgcn_s_wait_tensorcnt(0);
    }
    __syncthreads();

    // ---------------- P2: y = W2 @ feat + b2 -> Ybuf (fp32) ---------------
    {
        v16bf bfrag[8];
        const __bf16* bb = &Feat[(col0 + r) * LDK + 8 * half];
        #pragma unroll
        for (int k = 0; k < 8; ++k) {
            v8bf lo = *(const v8bf*)(bb + k * 32);
            v8bf hi = *(const v8bf*)(bb + k * 32 + 16);
            v16bf f;
            #pragma unroll
            for (int j = 0; j < 8; ++j) { f[j] = lo[j]; f[8 + j] = hi[j]; }
            bfrag[k] = f;
        }
        __syncthreads();   // all waves hold B frags before Ybuf overwrites Bmat/Feat
        for (int m = 0; m < 16; ++m) {
            v8f acc;
            float4 bl = *(const float4*)&b2[m * 16 + 8 * half];
            float4 bh = *(const float4*)&b2[m * 16 + 8 * half + 4];
            acc[0]=bl.x; acc[1]=bl.y; acc[2]=bl.z; acc[3]=bl.w;
            acc[4]=bh.x; acc[5]=bh.y; acc[6]=bh.z; acc[7]=bh.w;
            const __bf16* aa = &Wlds[(m * 16 + r) * LDK + 8 * half];
            #pragma unroll
            for (int k = 0; k < 8; ++k) {
                v8bf lo = *(const v8bf*)(aa + k * 32);
                v8bf hi = *(const v8bf*)(aa + k * 32 + 16);
                v16bf af;
                #pragma unroll
                for (int j = 0; j < 8; ++j) { af[j] = lo[j]; af[8 + j] = hi[j]; }
                acc = __builtin_amdgcn_wmma_f32_16x16x32_bf16(
                          false, af, false, bfrag[k], (short)0, acc, false, false);
            }
            v4f y0; y0[0]=acc[0]; y0[1]=acc[1]; y0[2]=acc[2]; y0[3]=acc[3];
            v4f y1; y1[0]=acc[4]; y1[1]=acc[5]; y1[2]=acc[6]; y1[3]=acc[7];
            float* yp = &Ybuf[(col0 + r) * LDY + m * 16 + 8 * half];
            *(v4f*)yp       = y0;
            *(v4f*)(yp + 4) = y1;
        }
    }
    __syncthreads();

    // ---------------- P3: IDWT + store ------------------------------------
    {
        float* ob = out + (size_t)n * (C64 * 128 * 128);
        #pragma unroll 2
        for (int it = 0; it < 32; ++it) {
            int i  = it * 256 + tid;             // 8192 = 64co * 128cols
            int co = i >> 7;
            int t  = i & 127;
            int p  = tile * 128 + t;
            int hh = p >> 6, ww = p & 63;
            const float* yp = &Ybuf[t * LDY + co];
            float y1 = 0.5f * yp[0];
            float y2 = 0.5f * yp[64];
            float y3 = 0.5f * yp[128];
            float y4 = 0.5f * yp[192];
            float a = y1 - y2 - y3 + y4;
            float b = y1 - y2 + y3 - y4;
            float c = y1 + y2 - y3 - y4;
            float d = y1 + y2 + y3 + y4;
            float* po = ob + ((size_t)co * 128 + 2 * hh) * 128 + 2 * ww;
            *(float2*)po         = make_float2(a, c);   // row 2hh
            *(float2*)(po + 128) = make_float2(b, d);   // row 2hh+1
        }
    }
}

extern "C" void kernel_launch(void* const* d_in, const int* in_sizes, int n_in,
                              void* d_out, int out_size, void* d_ws, size_t ws_size,
                              hipStream_t stream) {
    const float* x  = (const float*)d_in[0];
    const float* w1 = (const float*)d_in[1];   // conv_w     (256, 256) [out][in]
    const float* b1 = (const float*)d_in[2];   // conv_b     (256,)
    const float* w2 = (const float*)d_in[3];   // conv_out_w (256, 256) [out][in]
    const float* b2 = (const float*)d_in[4];   // conv_out_b (256,)
    float*   out = (float*)d_out;
    __bf16*  wbf = (__bf16*)d_ws;              // needs 262144 bytes of scratch

    hipFuncSetAttribute(reinterpret_cast<const void*>(wavelet_fused),
                        hipFuncAttributeMaxDynamicSharedMemorySize, LDS_BYTES);

    // 1) convert weights fp32 -> bf16 once per launch (512 KB, L2-resident)
    convert_weights<<<dim3(128), dim3(256), 0, stream>>>(w1, w2, wbf);

    // 2) fused DWT + GEMM1 + SiLU + GEMM2 + IDWT
    //    16 images * 32 column-tiles = 512 workgroups, 256 threads (8 waves)
    wavelet_fused<<<dim3(NIMG * 32), dim3(256), LDS_BYTES, stream>>>(
        x, wbf, b1, b2, out);
}
